// MSA_71047349010798
// MI455X (gfx1250) — matrix operand order repaired and checked by reading, not compiled, not run on hardware
//
#include <hip/hip_runtime.h>

typedef __attribute__((ext_vector_type(16))) _Float16 v16h;
typedef __attribute__((ext_vector_type(8)))  float    v8f;

#define B_SZ  2
#define S_LEN 4096
#define NH    8
#define HD    64
#define DM    512

#define KSTRIDE 72   // halfs per K-tile LDS row (144B: 16B-aligned, bank-spread)
#define VSTRIDE 40   // halfs per V-tile LDS row (80B: 16B-aligned, bank-spread)

union Frag {
    v16h v;
    _Float16 h[16];
    unsigned int u[8];
    uint4 q[2];
};

__device__ __forceinline__ v8f wmma_f16(v16h a, v16h b, v8f c) {
    // D = A(16x32 f16) * B(32x16 f16) + C(16x16 f32)
    return __builtin_amdgcn_wmma_f32_16x16x32_f16(
        /*neg_a=*/false, a, /*neg_b=*/false, b,
        /*c_mod=*/(short)0, c, /*reuse_a=*/false, /*reuse_b=*/false);
}

// ---- DPP16 16-lane butterflies: one VOP2+DPP instruction per stage -------
__device__ __forceinline__ float row16_max(float x) {
    float t;
    asm("v_max_num_f32_dpp %0, %1, %1 quad_perm:[1,0,3,2] row_mask:0xf bank_mask:0xf bound_ctrl:1" : "=v"(t) : "v"(x));
    asm("v_max_num_f32_dpp %0, %1, %1 quad_perm:[2,3,0,1] row_mask:0xf bank_mask:0xf bound_ctrl:1" : "=v"(x) : "v"(t));
    asm("v_max_num_f32_dpp %0, %1, %1 row_half_mirror row_mask:0xf bank_mask:0xf bound_ctrl:1"     : "=v"(t) : "v"(x));
    asm("v_max_num_f32_dpp %0, %1, %1 row_mirror row_mask:0xf bank_mask:0xf bound_ctrl:1"          : "=v"(x) : "v"(t));
    return x;
}
__device__ __forceinline__ float row16_sum(float x) {
    float t;
    asm("v_add_f32_dpp %0, %1, %1 quad_perm:[1,0,3,2] row_mask:0xf bank_mask:0xf bound_ctrl:1" : "=v"(t) : "v"(x));
    asm("v_add_f32_dpp %0, %1, %1 quad_perm:[2,3,0,1] row_mask:0xf bank_mask:0xf bound_ctrl:1" : "=v"(x) : "v"(t));
    asm("v_add_f32_dpp %0, %1, %1 row_half_mirror row_mask:0xf bank_mask:0xf bound_ctrl:1"     : "=v"(t) : "v"(x));
    asm("v_add_f32_dpp %0, %1, %1 row_mirror row_mask:0xf bank_mask:0xf bound_ctrl:1"          : "=v"(x) : "v"(t));
    return x;
}

// -------------------------------------------------------------------------
// Kernel 1: one per-head projection  y = x @ W^T + b  (f32 in -> f16 out).
// MODE 0: Q (scaled by 1/sqrt(512)), (b,h,s,64); MODE 1: K, (b,h,s,64);
// MODE 2: V, stored transposed (b,h,64,s).
// -------------------------------------------------------------------------
template <int MODE>
__global__ void msa_proj_kernel(const float* __restrict__ x_in,
                                const float* __restrict__ W,
                                const float* __restrict__ bias,
                                _Float16* __restrict__ outp) {
    const int wid  = blockIdx.x * (blockDim.x >> 5) + (threadIdx.x >> 5);
    const int lane = threadIdx.x & 31;
    const int half = lane >> 4;
    const int lm   = lane & 15;
    const int h    = wid & (NH - 1);
    const int tile = wid >> 3;
    const int b    = tile >> 8;            // S_LEN/16 == 256
    const int s0   = (tile & 255) * 16;

    const float* xrow = x_in + ((size_t)(b * S_LEN + s0 + lm)) * DM + h * HD;
    Frag a[2];
    #pragma unroll
    for (int c = 0; c < 2; ++c) {
        #pragma unroll
        for (int j = 0; j < 8; ++j) {
            const int k = c * 32 + 2 * (j & 3) + ((j >= 4) ? 16 : 0) + 8 * half;
            const float2 p = *(const float2*)(xrow + k);
            a[c].h[2 * j]     = (_Float16)p.x;
            a[c].h[2 * j + 1] = (_Float16)p.y;
        }
    }

    const float scale = 0.044194173824159216f;   // 1/sqrt(512)
    const size_t headbase = (size_t)(b * NH + h) * S_LEN;

    #pragma unroll 1
    for (int nt = 0; nt < 4; ++nt) {
        const int n = nt * 16 + lm;
        const float* wrow = W + (size_t)n * HD;
        Frag bf[2];
        #pragma unroll
        for (int c = 0; c < 2; ++c) {
            #pragma unroll
            for (int j = 0; j < 8; ++j) {
                const int k = c * 32 + 2 * (j & 3) + ((j >= 4) ? 16 : 0) + 8 * half;
                const float2 p = *(const float2*)(wrow + k);
                bf[c].h[2 * j]     = (_Float16)p.x;
                bf[c].h[2 * j + 1] = (_Float16)p.y;
            }
        }
        v8f acc = {};
        acc = wmma_f16(a[0].v, bf[0].v, acc);
        acc = wmma_f16(a[1].v, bf[1].v, acc);

        const float bn = bias[n];
        #pragma unroll
        for (int r = 0; r < 8; ++r) {
            const int s = s0 + r + 8 * half;
            const float y = acc[r] + bn;
            if (MODE == 0) {
                outp[(headbase + s) * HD + n] = (_Float16)(y * scale);
            } else if (MODE == 1) {
                outp[(headbase + s) * HD + n] = (_Float16)y;
            } else {
                outp[((size_t)(b * NH + h) * HD + n) * S_LEN + s] = (_Float16)y;
            }
        }
    }
}

// -------------------------------------------------------------------------
// Kernel 2: flash attention. 256-thread block = 8 waves; each wave owns a
// 16-query tile (128 queries/block). K/V tiles are cooperatively staged to
// LDS and software-pipelined: the next tile's global loads are issued
// before computing the current tile, hiding L2 latency behind 8 WMMAs.
// -------------------------------------------------------------------------
__global__ void msa_attn_kernel(const _Float16* __restrict__ qp,
                                const _Float16* __restrict__ kp,
                                const _Float16* __restrict__ vpT,
                                _Float16* __restrict__ attn) {
    __shared__ _Float16 klds[32 * KSTRIDE];   // K tile: [key][dim]
    __shared__ _Float16 vlds[64 * VSTRIDE];   // V tile: [dim][key]
    __shared__ _Float16 pbuf[8][16 * 32];     // per-wave P transpose bounce

    const int tid  = threadIdx.x;
    const int w    = tid >> 5;
    const int lane = tid & 31;
    const int half = lane >> 4;
    const int lm   = lane & 15;
    const int s0   = blockIdx.x * 128 + w * 16;
    const int bh   = blockIdx.y;
    const int b    = bh >> 3;
    const int h    = bh & 7;

    const _Float16* qbase = qp  + ((size_t)bh * S_LEN + s0) * HD;
    const _Float16* kbase = kp  + (size_t)bh * S_LEN * HD;
    const _Float16* vbase = vpT + (size_t)bh * HD * S_LEN;

    // Q A-fragments (held for the whole loop), pre-scaled by 1/sqrt(512)
    Frag aq[2];
    #pragma unroll
    for (int c = 0; c < 2; ++c) {
        aq[c].q[0] = *(const uint4*)(qbase + (size_t)lm * HD + c * 32 + 16 * half);
        aq[c].q[1] = *(const uint4*)(qbase + (size_t)lm * HD + c * 32 + 16 * half + 8);
    }

    v8f O[4] = {};
    float mrow[8], lrow[8];
    #pragma unroll
    for (int r = 0; r < 8; ++r) { mrow[r] = -1e30f; lrow[r] = 0.0f; }

    // cooperative staging (256 threads move 32x64 K + 64x32 V halfs)
    const int kkey = tid >> 3, kcol = (tid & 7) * 8;   // 8 halfs each
    const int vdim = tid >> 2, vcol = (tid & 3) * 8;   // 8 halfs each
    const _Float16* kgp = kbase + (size_t)kkey * HD + kcol;
    const _Float16* vgp = vbase + (size_t)vdim * S_LEN + vcol;
    uint4 kreg = *(const uint4*)kgp;     // prefetch tile 0
    uint4 vreg = *(const uint4*)vgp;

    #pragma unroll 1
    for (int kt = 0; kt < S_LEN; kt += 32) {
        __syncthreads();   // previous iteration's LDS reads are done
        *(uint4*)&klds[kkey * KSTRIDE + kcol] = kreg;
        *(uint4*)&vlds[vdim * VSTRIDE + vcol] = vreg;
        __syncthreads();   // staged K/V visible to all waves

        if (kt + 32 < S_LEN) {           // issue next tile's loads early
            kgp += (size_t)32 * HD;
            vgp += 32;
            kreg = *(const uint4*)kgp;
            vreg = *(const uint4*)vgp;
        }

        // K^T B-fragments from LDS: B[d][n] = K[t*16 + n][d]
        Frag bk[2][2];
        #pragma unroll
        for (int t = 0; t < 2; ++t) {
            const _Float16* krow = &klds[(t * 16 + lm) * KSTRIDE];
            #pragma unroll
            for (int c = 0; c < 2; ++c) {
                bk[t][c].q[0] = *(const uint4*)(krow + c * 32 + 16 * half);
                bk[t][c].q[1] = *(const uint4*)(krow + c * 32 + 16 * half + 8);
            }
        }
        v8f st0 = {}, st1 = {};
        st0 = wmma_f16(aq[0].v, bk[0][0].v, st0);
        st0 = wmma_f16(aq[1].v, bk[0][1].v, st0);
        st1 = wmma_f16(aq[0].v, bk[1][0].v, st1);
        st1 = wmma_f16(aq[1].v, bk[1][1].v, st1);

        // online softmax per row (row = r + 8*half)
        float alpha[8];
        #pragma unroll
        for (int r = 0; r < 8; ++r) {
            const float a = st0[r], c2 = st1[r];
            const float mx = row16_max(fmaxf(a, c2));
            const float mnew = fmaxf(mrow[r], mx);
            const float al = __expf(mrow[r] - mnew);
            const float p0 = __expf(a - mnew);
            const float p1 = __expf(c2 - mnew);
            const float rs = row16_sum(p0 + p1);
            lrow[r] = lrow[r] * al + rs;
            mrow[r] = mnew;
            alpha[r] = al;
            const int m = r + 8 * half;
            pbuf[w][m * 32 + lm]      = (_Float16)p0;   // keys kt..kt+15
            pbuf[w][m * 32 + 16 + lm] = (_Float16)p1;   // keys kt+16..kt+31
        }
        #pragma unroll
        for (int nt = 0; nt < 4; ++nt)
            #pragma unroll
            for (int r = 0; r < 8; ++r)
                O[nt][r] *= alpha[r];

        // per-wave LDS bounce: DS ops are in-order within a wave, only a
        // compiler scheduling barrier is needed (no block barrier)
        __builtin_amdgcn_wave_barrier();
        Frag ap;   // P as A fragment (16 x 32, rows = queries, K = keys)
        ap.q[0] = *(const uint4*)&pbuf[w][lm * 32 + 16 * half];
        ap.q[1] = *(const uint4*)&pbuf[w][lm * 32 + 16 * half + 8];
        __builtin_amdgcn_wave_barrier();

        // O += P * V from LDS V tile (dim-major: key pairs contiguous)
        #pragma unroll
        for (int nt = 0; nt < 4; ++nt) {
            Frag bv;
            const _Float16* vrow = &vlds[(nt * 16 + lm) * VSTRIDE];
            bv.q[0] = *(const uint4*)(vrow + 16 * half);
            bv.q[1] = *(const uint4*)(vrow + 16 * half + 8);
            O[nt] = wmma_f16(ap.v, bv.v, O[nt]);
        }
    }

    float inv[8];
    #pragma unroll
    for (int r = 0; r < 8; ++r) inv[r] = 1.0f / lrow[r];
    #pragma unroll
    for (int nt = 0; nt < 4; ++nt)
        #pragma unroll
        for (int r = 0; r < 8; ++r) {
            const int s = s0 + r + 8 * half;
            attn[((size_t)(b * S_LEN + s)) * DM + h * HD + nt * 16 + lm] =
                (_Float16)(O[nt][r] * inv[r]);
        }
}

// -------------------------------------------------------------------------
// Kernel 3a: one-shot fc_w f32 -> f16 conversion (512 KB, L2-resident).
// -------------------------------------------------------------------------
__global__ void msa_cvtw_kernel(const float* __restrict__ W,
                                _Float16* __restrict__ Wh) {
    const int stride = gridDim.x * blockDim.x;
    const float2* W2 = (const float2*)W;
    for (int i = blockIdx.x * blockDim.x + threadIdx.x; i < DM * DM / 2; i += stride) {
        const float2 p = W2[i];
        union { unsigned int u; _Float16 h[2]; } pk;
        pk.h[0] = (_Float16)p.x;
        pk.h[1] = (_Float16)p.y;
        *(unsigned int*)&Wh[2 * i] = pk.u;
    }
}

// -------------------------------------------------------------------------
// Kernel 3b: out = attn @ fc_w^T + fc_b (8192x512 @ 512x512), f32 out.
// One wave computes a 32x64 output tile: A fragments reused 4x, B 2x,
// 128 chained WMMAs per wave.
// -------------------------------------------------------------------------
__global__ void msa_fc_kernel(const _Float16* __restrict__ x,
                              const _Float16* __restrict__ Wh,
                              const float* __restrict__ bias,
                              float* __restrict__ out) {
    const int lane = threadIdx.x & 31;
    const int half = lane >> 4;
    const int lm   = lane & 15;
    const int mt   = blockIdx.x;   // 32-token tile (0..255)
    const int ng   = blockIdx.y;   // 64-output group (0..7)

    const _Float16* xrow0 = x + ((size_t)mt * 32 + lm) * DM;
    const _Float16* xrow1 = xrow0 + (size_t)16 * DM;

    v8f acc[2][4] = {};
    #pragma unroll 1
    for (int c = 0; c < DM; c += 32) {
        Frag a0, a1;
        a0.q[0] = *(const uint4*)(xrow0 + c + 8 * half);
        a0.q[1] = *(const uint4*)(xrow0 + c + 16 + 8 * half);
        a1.q[0] = *(const uint4*)(xrow1 + c + 8 * half);
        a1.q[1] = *(const uint4*)(xrow1 + c + 16 + 8 * half);
        #pragma unroll
        for (int nt = 0; nt < 4; ++nt) {
            const int n = ng * 64 + nt * 16 + lm;
            const _Float16* wrow = Wh + (size_t)n * DM + c + 8 * half;
            Frag bf;
            bf.q[0] = *(const uint4*)(wrow);
            bf.q[1] = *(const uint4*)(wrow + 16);
            acc[0][nt] = wmma_f16(a0.v, bf.v, acc[0][nt]);
            acc[1][nt] = wmma_f16(a1.v, bf.v, acc[1][nt]);
        }
    }
    #pragma unroll
    for (int nt = 0; nt < 4; ++nt) {
        const int n = ng * 64 + nt * 16 + lm;
        const float bn = bias[n];
        #pragma unroll
        for (int m2 = 0; m2 < 2; ++m2)
            #pragma unroll
            for (int r = 0; r < 8; ++r) {
                const int tok = mt * 32 + m2 * 16 + r + 8 * half;
                out[(size_t)tok * DM + n] = acc[m2][nt][r] + bn;
            }
    }
}

extern "C" void kernel_launch(void* const* d_in, const int* in_sizes, int n_in,
                              void* d_out, int out_size, void* d_ws, size_t ws_size,
                              hipStream_t stream) {
    (void)in_sizes; (void)n_in; (void)out_size; (void)ws_size;
    const float* q   = (const float*)d_in[0];
    const float* k   = (const float*)d_in[1];
    const float* v   = (const float*)d_in[2];
    const float* Wq  = (const float*)d_in[3];
    const float* bq  = (const float*)d_in[4];
    const float* Wk  = (const float*)d_in[5];
    const float* bk  = (const float*)d_in[6];
    const float* Wv  = (const float*)d_in[7];
    const float* bv  = (const float*)d_in[8];
    const float* fcw = (const float*)d_in[9];
    const float* fcb = (const float*)d_in[10];

    char* ws = (char*)d_ws;
    const size_t per = (size_t)B_SZ * NH * S_LEN * HD * 2;  // 8 MiB each (f16)
    _Float16* qp   = (_Float16*)(ws);
    _Float16* kp   = (_Float16*)(ws + per);
    _Float16* vpT  = (_Float16*)(ws + 2 * per);
    _Float16* attn = (_Float16*)(ws + 3 * per);
    _Float16* wh   = (_Float16*)(ws + 4 * per);             // 512 KiB

    // 1) projections: 4096 waves per launch, 4 waves per block, no spills
    const dim3 pgrid((B_SZ * (S_LEN / 16) * NH) / 4);
    const dim3 pblk(128);
    msa_proj_kernel<0><<<pgrid, pblk, 0, stream>>>(q, Wq, bq, qp);
    msa_proj_kernel<1><<<pgrid, pblk, 0, stream>>>(k, Wk, bk, kp);
    msa_proj_kernel<2><<<pgrid, pblk, 0, stream>>>(v, Wv, bv, vpT);
    msa_cvtw_kernel<<<dim3(256), dim3(256), 0, stream>>>(fcw, wh);

    // 2) flash attention: 8 waves/block share LDS-staged, pipelined K/V
    msa_attn_kernel<<<dim3(S_LEN / 128, B_SZ * NH), dim3(256), 0, stream>>>(
        qp, kp, vpT, attn);

    // 3) output projection: 32x64 tile per wave
    msa_fc_kernel<<<dim3((B_SZ * S_LEN) / 32, DM / 64), dim3(32), 0, stream>>>(
        attn, wh, fcb, (float*)d_out);
}
